// DecoderWithAttention_26980984554007
// MI455X (gfx1250) — matrix operand order repaired and checked by reading, not compiled, not run on hardware
//
#include <hip/hip_runtime.h>
#include <hip/hip_bf16.h>
#include <stdint.h>

// Problem constants
#define BB 16
#define PP 196
#define EE 1024
#define AA 1024
#define DD 512
#define EMB 512
#define VV 10000
#define LL 32
#define TT 31
#define MROWS (BB * PP)        // 3136
#define KCAT 2048              // [x(1536) | h(512)]
#define NCAT 2048              // 4*D gates
#define NHG  2048              // [att2(1024) | gate(1024)]

// Output layout (floats, concatenated in reference return order)
#define OUT_PREDS 0
#define OUT_CAPS  (BB * TT * VV)                 // 4,960,000
#define OUT_DLEN  (OUT_CAPS + BB * LL)           // 4,960,512
#define OUT_ALPH  (OUT_DLEN + BB)                // 4,960,528
#define OUT_SIND  (OUT_ALPH + BB * TT * PP)      // 5,057,744

typedef __attribute__((ext_vector_type(16))) __bf16 v16bf;
typedef __attribute__((ext_vector_type(8)))  float  v8f;

union FragU {
    v16bf v;
    uint4 q[2];
};

__device__ __forceinline__ unsigned short f2bf(float f) {
    unsigned int u = __float_as_uint(f);
    unsigned int r = u + 0x7fffu + ((u >> 16) & 1u);   // round-to-nearest-even
    return (unsigned short)(r >> 16);
}
__device__ __forceinline__ float bf2f(unsigned short u) {
    return __uint_as_float(((unsigned int)u) << 16);
}

// A fragment (16x32 bf16 tile) from LDS, row-major 16 x ldk.
__device__ __forceinline__ v16bf load_a_frag(const unsigned short* smem, int ldk,
                                             int k0, int lane) {
    int row = lane & 15;
    int hi  = (lane >> 4) & 1;
    FragU f;
    f.q[0] = *(const uint4*)(smem + row * ldk + k0 + 8 * hi);
    f.q[1] = *(const uint4*)(smem + row * ldk + k0 + 16 + 8 * hi);
    return f.v;
}

// B fragment (32x16 bf16 tile) from W^T (N x K row-major bf16 in global/L2).
__device__ __forceinline__ v16bf load_b_frag(const unsigned short* wT, int K,
                                             int n0, int k0, int lane) {
    int col = n0 + (lane & 15);
    int hi  = (lane >> 4) & 1;
    const unsigned short* p = wT + (size_t)col * K + k0 + 8 * hi;
    FragU f;
    f.q[0] = ((const uint4*)p)[0];
    f.q[1] = ((const uint4*)p)[2];
    return f.v;
}

// ---------------- setup kernels ----------------

__global__ __launch_bounds__(512) void k_sort(const int* __restrict__ cap_len,
                                              const int* __restrict__ captions,
                                              int* __restrict__ sind,
                                              int* __restrict__ dlen,
                                              int* __restrict__ caps_s,
                                              float* __restrict__ out) {
    __shared__ int s_len[BB];
    __shared__ int s_ind[BB];
    int tid = threadIdx.x;
    if (tid < BB) s_len[tid] = cap_len[tid];
    __syncthreads();
    if (tid < BB) {
        int li = s_len[tid], rank = 0;
        for (int j = 0; j < BB; ++j) {
            int lj = s_len[j];
            if (lj > li || (lj == li && j < tid)) rank++;
        }
        s_ind[rank] = tid;   // stable descending argsort
    }
    __syncthreads();
    if (tid < BB) {
        int src = s_ind[tid];
        sind[tid] = src;
        int dl = s_len[src] - 1;
        dlen[tid] = dl;
        out[OUT_DLEN + tid] = (float)dl;
        out[OUT_SIND + tid] = (float)src;
    }
    __syncthreads();
    int r = tid >> 5, l = tid & 31;                     // 16 x 32 sorted captions
    int c = captions[s_ind[r] * LL + l];
    caps_s[r * LL + l] = c;
    out[OUT_CAPS + r * LL + l] = (float)c;
}

__global__ __launch_bounds__(256) void k_gather_enc(const float* __restrict__ enc,
                                                    const int* __restrict__ sind,
                                                    unsigned short* __restrict__ enc_bf) {
    int idx = blockIdx.x * 256 + threadIdx.x;           // over B*P*E
    int b = idx / (PP * EE);
    int rem = idx - b * (PP * EE);
    float v = enc[(size_t)sind[b] * PP * EE + rem];
    enc_bf[idx] = f2bf(v);
}

__global__ __launch_bounds__(256) void k_mean(const float* __restrict__ enc,
                                              const int* __restrict__ sind,
                                              float* __restrict__ mean) {
    int idx = blockIdx.x * 256 + threadIdx.x;           // over B*E
    int b = idx >> 10, e = idx & (EE - 1);
    const float* base = enc + (size_t)sind[b] * PP * EE + e;
    float s = 0.f;
    for (int p = 0; p < PP; ++p) s += base[(size_t)p * EE];
    mean[idx] = s * (1.0f / PP);
}

__global__ __launch_bounds__(256) void k_init_hc(const float* __restrict__ mean,
                                                 const float* __restrict__ hW,
                                                 const float* __restrict__ hb,
                                                 const float* __restrict__ cW,
                                                 const float* __restrict__ cb,
                                                 unsigned short* __restrict__ h_bf,
                                                 float* __restrict__ c) {
    int idx = blockIdx.x * 256 + threadIdx.x;           // over 2*B*D
    int which = idx / (BB * DD);
    int rem = idx - which * (BB * DD);
    int b = rem / DD, d = rem - b * DD;
    const float* W = which ? cW : hW;
    float s = 0.f;
    const float* m = mean + b * EE;
    for (int e = 0; e < EE; ++e) s += m[e] * W[e * DD + d];
    if (which) c[rem] = s + cb[d];
    else       h_bf[rem] = f2bf(s + hb[d]);
}

// W (K x N fp32, row-major) -> W^T (N x K bf16, row-major)
__global__ __launch_bounds__(256) void k_transpose(const float* __restrict__ W,
                                                   int K, int N,
                                                   unsigned short* __restrict__ WT) {
    long idx = (long)blockIdx.x * 256 + threadIdx.x;
    if (idx >= (long)K * N) return;
    int n = (int)(idx / K), k = (int)(idx - (long)n * K);
    WT[idx] = f2bf(W[(size_t)k * N + n]);
}

// Combined [W_ih ; W_hh] -> (2048 x 2048) bf16 transposed
__global__ __launch_bounds__(256) void k_wcat(const float* __restrict__ Wih,
                                              const float* __restrict__ Whh,
                                              unsigned short* __restrict__ WT) {
    long idx = (long)blockIdx.x * 256 + threadIdx.x;    // over 2048*2048
    int n = (int)(idx >> 11), k = (int)(idx & 2047);
    float v = (k < 1536) ? Wih[(size_t)k * NCAT + n]
                         : Whh[(size_t)(k - 1536) * NCAT + n];
    WT[idx] = f2bf(v);
}

// Combined [dec_att_W | f_beta_W] -> (2048 x 512) bf16 transposed
__global__ __launch_bounds__(256) void k_whg(const float* __restrict__ decW,
                                             const float* __restrict__ fbW,
                                             unsigned short* __restrict__ WT) {
    long idx = (long)blockIdx.x * 256 + threadIdx.x;    // over 2048*512
    int n = (int)(idx >> 9), k = (int)(idx & (DD - 1));
    float v = (n < AA) ? decW[(size_t)k * AA + n]
                       : fbW[(size_t)k * EE + (n - AA)];
    WT[idx] = f2bf(v);
}

// ---------------- WMMA GEMM: att1 = enc_bf(3136x1024) @ enc_att_W + b (bf16 out) ----------------

__global__ __launch_bounds__(256) void k_att1(const unsigned short* __restrict__ enc_bf,
                                              const unsigned short* __restrict__ wT,
                                              const float* __restrict__ bias,
                                              unsigned short* __restrict__ att1_bf) {
    __shared__ unsigned short As[16 * AA];              // 32 KB
    int mtile = blockIdx.x >> 3;                        // 196 tiles of M
    int ngrp  = blockIdx.x & 7;                         // 8 groups x 8 waves = 64 N tiles
    const uint4* src = (const uint4*)(enc_bf + (size_t)mtile * 16 * AA);
    for (int i = threadIdx.x; i < 16 * AA / 8; i += 256) ((uint4*)As)[i] = src[i];
    __syncthreads();
    int wave = threadIdx.x >> 5, lane = threadIdx.x & 31;
    int n0 = (ngrp * 8 + wave) * 16;
    v8f acc = {0.f, 0.f, 0.f, 0.f, 0.f, 0.f, 0.f, 0.f};
    for (int k0 = 0; k0 < AA; k0 += 32) {
        v16bf a = load_a_frag(As, AA, k0, lane);
        v16bf b = load_b_frag(wT, AA, n0, k0, lane);
        acc = __builtin_amdgcn_wmma_f32_16x16x32_bf16(false, a, false, b,
                                                      (short)0, acc, false, false);
    }
    int n = n0 + (lane & 15);
    int mbase = mtile * 16 + ((lane >> 4) & 1) * 8;
    float bn = bias[n];
    for (int r = 0; r < 8; ++r)
        att1_bf[(size_t)(mbase + r) * AA + n] = f2bf(acc[r] + bn);
}

// ---------------- per-step kernels ----------------

// WMMA: [att2 | gate_lin] = h_bf(16x512) @ Whg^T(2048x512); split epilogue
__global__ __launch_bounds__(256) void k_hgemv(const unsigned short* __restrict__ hbf,
                                               const unsigned short* __restrict__ wT,
                                               const float* __restrict__ decB,
                                               const float* __restrict__ fbB,
                                               float* __restrict__ att2,
                                               float* __restrict__ gate) {
    __shared__ unsigned short As[16 * DD];              // 16 KB
    for (int i = threadIdx.x; i < 16 * DD / 8; i += 256)
        ((uint4*)As)[i] = ((const uint4*)hbf)[i];
    __syncthreads();
    int wave = threadIdx.x >> 5, lane = threadIdx.x & 31;
    int n0 = (blockIdx.x * 8 + wave) * 16;              // 128 N tiles, 16 blocks
    v8f acc = {0.f, 0.f, 0.f, 0.f, 0.f, 0.f, 0.f, 0.f};
    for (int k0 = 0; k0 < DD; k0 += 32) {
        v16bf a = load_a_frag(As, DD, k0, lane);
        v16bf b = load_b_frag(wT, DD, n0, k0, lane);
        acc = __builtin_amdgcn_wmma_f32_16x16x32_bf16(false, a, false, b,
                                                      (short)0, acc, false, false);
    }
    int n = n0 + (lane & 15);
    int mbase = ((lane >> 4) & 1) * 8;
    if (n0 < AA) {                                      // wave-uniform split
        float bn = decB[n];
        for (int r = 0; r < 8; ++r)
            att2[(size_t)(mbase + r) * AA + n] = acc[r] + bn;
    } else {
        int nn = n - AA;
        float bn = fbB[nn];
        for (int r = 0; r < 8; ++r) {
            float s = acc[r] + bn;
            gate[(size_t)(mbase + r) * EE + nn] = 1.0f / (1.0f + __expf(-s));
        }
    }
}

// scores[b,p] = sum_a relu(att1_bf + att2) * fw + fb
__global__ __launch_bounds__(256) void k_scores(const unsigned short* __restrict__ att1_bf,
                                                const float* __restrict__ att2,
                                                const float* __restrict__ fw,
                                                const float* __restrict__ fb,
                                                float* __restrict__ scores) {
    __shared__ float red[256];
    int row = blockIdx.x;                               // B*P rows
    int b = row / PP;
    int tid = threadIdx.x;
    const unsigned short* a1 = att1_bf + (size_t)row * AA;
    const float* a2 = att2 + b * AA;
    float s = 0.f;
    for (int a = tid; a < AA; a += 256) {
        float r = bf2f(a1[a]) + a2[a];
        r = r > 0.f ? r : 0.f;
        s += r * fw[a];
    }
    red[tid] = s;
    __syncthreads();
    for (int st = 128; st > 0; st >>= 1) {
        if (tid < st) red[tid] += red[tid + st];
        __syncthreads();
    }
    if (tid == 0) scores[row] = red[0] + fb[0];
}

__global__ __launch_bounds__(256) void k_softmax(const float* __restrict__ scores,
                                                 const int* __restrict__ dlen,
                                                 float* __restrict__ alpha,
                                                 float* __restrict__ out, int t) {
    __shared__ float red[256];
    int b = blockIdx.x, tid = threadIdx.x;
    float v = (tid < PP) ? scores[b * PP + tid] : -1e30f;
    red[tid] = v;
    __syncthreads();
    for (int st = 128; st > 0; st >>= 1) {
        if (tid < st) red[tid] = fmaxf(red[tid], red[tid + st]);
        __syncthreads();
    }
    float m = red[0];
    __syncthreads();
    float e = (tid < PP) ? __expf(v - m) : 0.f;
    red[tid] = e;
    __syncthreads();
    for (int st = 128; st > 0; st >>= 1) {
        if (tid < st) red[tid] += red[tid + st];
        __syncthreads();
    }
    float inv = 1.0f / red[0];
    if (tid < PP) {
        float al = e * inv;
        alpha[b * PP + tid] = al;
        int active = dlen[b] > t;
        out[OUT_ALPH + b * (TT * PP) + t * PP + tid] = active ? al : 0.f;
    }
}

// build x_bf = [emb_t | gate+awe | h] (16 x 2048 bf16); awe from sorted bf16 encoder
__global__ __launch_bounds__(256) void k_awex(const unsigned short* __restrict__ enc_bf,
                                              const float* __restrict__ alpha,
                                              const float* __restrict__ gate,
                                              const float* __restrict__ emb,
                                              const int* __restrict__ caps_s,
                                              const unsigned short* __restrict__ hbf,
                                              unsigned short* __restrict__ xbf, int t) {
    int idx = blockIdx.x * 256 + threadIdx.x;           // over B*2048
    int b = idx >> 11, j = idx & 2047;
    if (j < EMB) {
        int cap = caps_s[b * LL + t];
        xbf[idx] = f2bf(emb[(size_t)cap * EMB + j]);
    } else if (j < EMB + EE) {
        int e = j - EMB;
        const unsigned short* base = enc_bf + (size_t)b * PP * EE + e;
        const float* al = alpha + b * PP;
        float s = 0.f;
        for (int p = 0; p < PP; ++p) s += bf2f(base[(size_t)p * EE]) * al[p];
        xbf[idx] = f2bf(gate[b * EE + e] + s);
    } else {
        xbf[idx] = hbf[b * DD + (j - EMB - EE)];
    }
}

// gates = [x|h](16x2048) @ Wcat^T(2048x2048) + b_ih + b_hh
__global__ __launch_bounds__(256) void k_gates(const unsigned short* __restrict__ xbf,
                                               const unsigned short* __restrict__ wT,
                                               const float* __restrict__ bih,
                                               const float* __restrict__ bhh,
                                               float* __restrict__ gates) {
    __shared__ unsigned short As[16 * KCAT];            // 64 KB
    for (int i = threadIdx.x; i < 16 * KCAT / 8; i += 256)
        ((uint4*)As)[i] = ((const uint4*)xbf)[i];
    __syncthreads();
    int wave = threadIdx.x >> 5, lane = threadIdx.x & 31;
    int n0 = (blockIdx.x * 8 + wave) * 16;              // 128 N tiles, 16 blocks
    v8f acc = {0.f, 0.f, 0.f, 0.f, 0.f, 0.f, 0.f, 0.f};
    for (int k0 = 0; k0 < KCAT; k0 += 32) {
        v16bf a = load_a_frag(As, KCAT, k0, lane);
        v16bf b = load_b_frag(wT, KCAT, n0, k0, lane);
        acc = __builtin_amdgcn_wmma_f32_16x16x32_bf16(false, a, false, b,
                                                      (short)0, acc, false, false);
    }
    int n = n0 + (lane & 15);
    int mbase = ((lane >> 4) & 1) * 8;
    float bn = bih[n] + bhh[n];
    for (int r = 0; r < 8; ++r)
        gates[(size_t)(mbase + r) * NCAT + n] = acc[r] + bn;
}

__global__ __launch_bounds__(256) void k_lstm(const float* __restrict__ gates,
                                              unsigned short* __restrict__ h_bf,
                                              float* __restrict__ c,
                                              const int* __restrict__ dlen,
                                              unsigned short* __restrict__ hnew_bf, int t) {
    int idx = blockIdx.x * 256 + threadIdx.x;           // over B*D
    int b = idx >> 9, d = idx & (DD - 1);
    const float* g = gates + (size_t)b * NCAT;
    float gi = g[d], gf = g[DD + d], gg = g[2 * DD + d], go = g[3 * DD + d];
    float si = 1.0f / (1.0f + __expf(-gi));
    float sf = 1.0f / (1.0f + __expf(-gf));
    float so = 1.0f / (1.0f + __expf(-go));
    float cn = sf * c[idx] + si * tanhf(gg);
    float hn = so * tanhf(cn);
    int active = dlen[b] > t;
    if (active) { h_bf[idx] = f2bf(hn); c[idx] = cn; }
    hnew_bf[idx] = f2bf(hn);                            // preds use h_new regardless
}

// preds = h_new(16x512) @ fc_W^T(10000x512) + fc_b, masked by active
__global__ __launch_bounds__(256) void k_fc(const unsigned short* __restrict__ hbf,
                                            const unsigned short* __restrict__ wT,
                                            const float* __restrict__ bias,
                                            const int* __restrict__ dlen,
                                            float* __restrict__ out, int t) {
    __shared__ unsigned short As[16 * DD];              // 16 KB
    for (int i = threadIdx.x; i < 16 * DD / 8; i += 256)
        ((uint4*)As)[i] = ((const uint4*)hbf)[i];
    __syncthreads();
    int wave = threadIdx.x >> 5, lane = threadIdx.x & 31;
    int tile = blockIdx.x * 8 + wave;                   // 625 tiles of N
    if (tile >= VV / 16) return;                        // wave-uniform guard
    int n0 = tile * 16;
    v8f acc = {0.f, 0.f, 0.f, 0.f, 0.f, 0.f, 0.f, 0.f};
    for (int k0 = 0; k0 < DD; k0 += 32) {
        v16bf a = load_a_frag(As, DD, k0, lane);
        v16bf b = load_b_frag(wT, DD, n0, k0, lane);
        acc = __builtin_amdgcn_wmma_f32_16x16x32_bf16(false, a, false, b,
                                                      (short)0, acc, false, false);
    }
    int n = n0 + (lane & 15);
    int mbase = ((lane >> 4) & 1) * 8;
    float bn = bias[n];
    for (int r = 0; r < 8; ++r) {
        int m = mbase + r;
        float v = (dlen[m] > t) ? (acc[r] + bn) : 0.f;
        out[OUT_PREDS + (size_t)m * (TT * VV) + (size_t)t * VV + n] = v;
    }
}

// ---------------- host launcher ----------------

extern "C" void kernel_launch(void* const* d_in, const int* in_sizes, int n_in,
                              void* d_out, int out_size, void* d_ws, size_t ws_size,
                              hipStream_t stream) {
    const float* encoder_out = (const float*)d_in[0];
    const float* enc_att_W   = (const float*)d_in[1];
    const float* enc_att_b   = (const float*)d_in[2];
    const float* dec_att_W   = (const float*)d_in[3];
    const float* dec_att_b   = (const float*)d_in[4];
    const float* full_att_W  = (const float*)d_in[5];
    const float* full_att_b  = (const float*)d_in[6];
    const float* emb         = (const float*)d_in[7];
    const float* W_ih        = (const float*)d_in[8];
    const float* b_ih        = (const float*)d_in[9];
    const float* W_hh        = (const float*)d_in[10];
    const float* b_hh        = (const float*)d_in[11];
    const float* init_h_W    = (const float*)d_in[12];
    const float* init_h_b    = (const float*)d_in[13];
    const float* init_c_W    = (const float*)d_in[14];
    const float* init_c_b    = (const float*)d_in[15];
    const float* f_beta_W    = (const float*)d_in[16];
    const float* f_beta_b    = (const float*)d_in[17];
    const float* fc_W        = (const float*)d_in[18];
    const float* fc_b        = (const float*)d_in[19];
    const int*   captions    = (const int*)d_in[20];
    const int*   cap_len     = (const int*)d_in[21];
    float* out = (float*)d_out;

    char* ws = (char*)d_ws;
    size_t off = 0;
    auto walloc = [&](size_t bytes) -> void* {
        void* p = ws + off;
        off = (off + bytes + 255) & ~(size_t)255;
        return p;
    };
    unsigned short* wcatT    = (unsigned short*)walloc((size_t)NCAT * KCAT * 2);
    unsigned short* fcWT     = (unsigned short*)walloc((size_t)VV * DD * 2);
    unsigned short* encAttWT = (unsigned short*)walloc((size_t)AA * AA * 2);
    unsigned short* whgT     = (unsigned short*)walloc((size_t)NHG * DD * 2);
    unsigned short* enc_bf   = (unsigned short*)walloc((size_t)MROWS * EE * 2);
    unsigned short* att1_bf  = (unsigned short*)walloc((size_t)MROWS * AA * 2);
    float* meanb  = (float*)walloc((size_t)BB * EE * 4);
    float* c      = (float*)walloc((size_t)BB * DD * 4);
    float* att2   = (float*)walloc((size_t)BB * AA * 4);
    float* gate   = (float*)walloc((size_t)BB * EE * 4);
    float* scores = (float*)walloc((size_t)BB * PP * 4);
    float* alpha  = (float*)walloc((size_t)BB * PP * 4);
    unsigned short* h_bf    = (unsigned short*)walloc((size_t)BB * DD * 2);
    unsigned short* xbf     = (unsigned short*)walloc((size_t)BB * KCAT * 2);
    unsigned short* hnew_bf = (unsigned short*)walloc((size_t)BB * DD * 2);
    float* gatesb = (float*)walloc((size_t)BB * NCAT * 4);
    int* sind   = (int*)walloc(BB * 4);
    int* dlen   = (int*)walloc(BB * 4);
    int* caps_s = (int*)walloc(BB * LL * 4);

    // setup
    k_sort<<<1, 512, 0, stream>>>(cap_len, captions, sind, dlen, caps_s, out);
    k_gather_enc<<<(MROWS * EE) / 256, 256, 0, stream>>>(encoder_out, sind, enc_bf);
    k_mean<<<(BB * EE) / 256, 256, 0, stream>>>(encoder_out, sind, meanb);
    k_init_hc<<<(2 * BB * DD) / 256, 256, 0, stream>>>(meanb, init_h_W, init_h_b,
                                                       init_c_W, init_c_b, h_bf, c);
    k_transpose<<<(AA * AA) / 256, 256, 0, stream>>>(enc_att_W, AA, AA, encAttWT);
    k_transpose<<<(DD * VV) / 256, 256, 0, stream>>>(fc_W, DD, VV, fcWT);
    k_wcat<<<(NCAT * KCAT) / 256, 256, 0, stream>>>(W_ih, W_hh, wcatT);
    k_whg<<<(NHG * DD) / 256, 256, 0, stream>>>(dec_att_W, f_beta_W, whgT);
    k_att1<<<PP * 8, 256, 0, stream>>>(enc_bf, encAttWT, enc_att_b, att1_bf);

    // 31 serial decode steps
    for (int t = 0; t < TT; ++t) {
        k_hgemv<<<NHG / 128, 256, 0, stream>>>(h_bf, whgT, dec_att_b, f_beta_b,
                                               att2, gate);
        k_scores<<<MROWS, 256, 0, stream>>>(att1_bf, att2, full_att_W, full_att_b,
                                            scores);
        k_softmax<<<BB, 256, 0, stream>>>(scores, dlen, alpha, out, t);
        k_awex<<<(BB * KCAT) / 256, 256, 0, stream>>>(enc_bf, alpha, gate, emb,
                                                      caps_s, h_bf, xbf, t);
        k_gates<<<NCAT / 128, 256, 0, stream>>>(xbf, wcatT, b_ih, b_hh, gatesb);
        k_lstm<<<(BB * DD) / 256, 256, 0, stream>>>(gatesb, h_bf, c, dlen, hnew_bf, t);
        k_fc<<<(VV / 16 + 7) / 8, 256, 0, stream>>>(hnew_bf, fcWT, fc_b, dlen, out, t);
    }
}
